// EncodingModel_34634616275399
// MI455X (gfx1250) — matrix operand order repaired and checked by reading, not compiled, not run on hardware
//
#include <hip/hip_runtime.h>
#include <hip/hip_bf16.h>
#include <stdint.h>

// ---------------- problem constants ----------------
#define N_NODES   50000
#define N_PHRASES 1024
#define DIM       256
#define RNB       4
#define NEG_INF   (-999999.0f)

// Pad node dim so it divides evenly into 128-wide block tiles:
// 50048 = 128 * 391
#define NPAD      50048
#define NGROUPS   (NPAD / 128)           // 391 N-blocks
#define MBLOCKS   (N_PHRASES / 128)      // 8 M-blocks

// LDS B-tile row stride in f16 elements: 128 K-halves + 8 pad.
// stride = 136 f16 = 272 B = 68 dwords; 68 % 64 = 4 -> consecutive rows sit
// 4 banks apart, so lanes 0..15 (rows r..r+15) cover all 64 banks
// conflict-free for 16B ds loads; 272 % 16 == 0 keeps b128 alignment.
#define BSTRIDE   136

typedef __attribute__((ext_vector_type(16))) _Float16 v16h;
typedef __attribute__((ext_vector_type(8)))  _Float16 v8h;
typedef __attribute__((ext_vector_type(8)))  float    v8f;

// =====================================================================
// Kernel 1: phrase prep. One wave32 per phrase row.
//   pF16[m][d] = (f16) normalize(P[m])[d]
//   pwBias[m]  = normalize(P[m]) . w_p + score_b
// =====================================================================
__global__ __launch_bounds__(256) void phrase_prep_kernel(
    const float* __restrict__ P, const float* __restrict__ score_w,
    const float* __restrict__ score_b,
    _Float16* __restrict__ pF16, float* __restrict__ pwBias)
{
    const int lane = threadIdx.x & 31;
    const int row  = blockIdx.x * 8 + (threadIdx.x >> 5);
    if (row >= N_PHRASES) return;

    const float* src = P + (size_t)row * DIM + lane * 8;
    float x[8];
    float ss = 0.0f;
#pragma unroll
    for (int i = 0; i < 8; ++i) { x[i] = src[i]; ss += x[i] * x[i]; }
#pragma unroll
    for (int m = 16; m >= 1; m >>= 1) ss += __shfl_xor(ss, m, 32);

    const float inv = 1.0f / fmaxf(sqrtf(ss), 1e-8f);

    const float* wp  = score_w + lane * 8;           // w_p = score_w[0:256]
    _Float16*    dst = pF16 + (size_t)row * DIM + lane * 8;
    float dot = 0.0f;
#pragma unroll
    for (int i = 0; i < 8; ++i) {
        const float v = x[i] * inv;
        dst[i] = (_Float16)v;
        dot += v * wp[i];
    }
#pragma unroll
    for (int m = 16; m >= 1; m >>= 1) dot += __shfl_xor(dot, m, 32);

    if (lane == 0) pwBias[row] = dot + score_b[0];
}

// =====================================================================
// Kernel 2a: per-node inverse L2 norm. One wave32 per node row.
// =====================================================================
__global__ __launch_bounds__(256) void node_norm_kernel(
    const float* __restrict__ NE, float* __restrict__ inv_norm)
{
    const int lane = threadIdx.x & 31;
    const int row  = blockIdx.x * 8 + (threadIdx.x >> 5);
    if (row >= N_NODES) return;

    const float* src = NE + (size_t)row * DIM + lane * 8;
    float ss = 0.0f;
#pragma unroll
    for (int i = 0; i < 8; ++i) { const float v = src[i]; ss += v * v; }
#pragma unroll
    for (int m = 16; m >= 1; m >>= 1) ss += __shfl_xor(ss, m, 32);

    if (lane == 0) inv_norm[row] = 1.0f / fmaxf(sqrtf(ss), 1e-8f);
}

// =====================================================================
// Kernel 2b: fused node features. One wave32 per node (padded range).
//   crossF16[n][d] = (f16) sum_k feats[k][d] * w_c[k][d]
//   nodeBias[n]    = sum_{k,d} feats[k][d]*w_n[k][d] + (filter ? 0 : NEG_INF)
// Rows n in [N_NODES, NPAD) are zero-filled so the GEMM needs no bounds
// checks around its WMMA loop.
// =====================================================================
__global__ __launch_bounds__(256) void node_feats_kernel(
    const float* __restrict__ NE, const float* __restrict__ score_w,
    const long long* __restrict__ neighbors, const int* __restrict__ neighbor_mask,
    const int* __restrict__ node_filter_mask, const float* __restrict__ inv_norm,
    _Float16* __restrict__ crossF16, float* __restrict__ nodeBias)
{
    const int lane = threadIdx.x & 31;
    const int node = blockIdx.x * 8 + (threadIdx.x >> 5);
    if (node >= NPAD) return;

    const int d0 = lane * 8;
    _Float16* dst = crossF16 + (size_t)node * DIM + d0;

    if (node >= N_NODES) {           // zero padding rows
#pragma unroll
        for (int i = 0; i < 8; ++i) dst[i] = (_Float16)0.0f;
        if (lane == 0) nodeBias[node] = 0.0f;
        return;
    }

    // self contribution (k = 0)
    const float invs = inv_norm[node];
    const float* self = NE + (size_t)node * DIM + d0;
    const float* wn0  = score_w + DIM + d0;          // w_n[0]
    const float* wc0  = wn0 + DIM;                   // w_c[0]

    float cr[8];
    float scal = 0.0f;
#pragma unroll
    for (int i = 0; i < 8; ++i) {
        const float v = self[i] * invs;
        cr[i] = v * wc0[i];
        scal += v * wn0[i];
    }

    // neighbor contributions (k = 1..4)
#pragma unroll
    for (int r = 0; r < RNB; ++r) {
        const int msk = neighbor_mask[(size_t)node * RNB + r];
        if (msk != 0) {
            const long long nb = neighbors[(size_t)node * RNB + r];
            const float invr = inv_norm[(int)nb];
            const float* emb = NE + (size_t)nb * DIM + d0;
            const float* wnk = score_w + DIM + (size_t)(r + 1) * (2 * DIM) + d0;
            const float* wck = wnk + DIM;
#pragma unroll
            for (int i = 0; i < 8; ++i) {
                const float v = emb[i] * invr;
                cr[i] += v * wck[i];
                scal  += v * wnk[i];
            }
        }
    }

#pragma unroll
    for (int m = 16; m >= 1; m >>= 1) scal += __shfl_xor(scal, m, 32);

#pragma unroll
    for (int i = 0; i < 8; ++i) dst[i] = (_Float16)cr[i];

    if (lane == 0) {
        const float lm = (node_filter_mask[node] > 0) ? 0.0f : NEG_INF;
        nodeBias[node] = scal + lm;
    }
}

// =====================================================================
// Kernel 3: WMMA GEMM with LDS-shared B.
//   logits[m][n] = pF16[m,:] . crossF16[n,:] + pwBias[m] + nodeBias[n]
// M = 1024, N = NPAD, K = 256.
// Block = 8 waves = 128 M-rows x 128 N-cols. The 8 waves share the B tile
// (128 nodes x 256 K), staged through LDS in two K-halves of 34 KB, so B
// is read from L2 once per block (8x less L2 traffic than per-wave reads).
// Each wave: 16x128 output tile, 8 f32 accumulators, 64 WMMA total.
//
// VGPR layouts per CDNA5 ISA 7.12.2 (wave32):
//  A 16x32 f16: lane L -> row m0+(L&15); halves K = kb..kb+7, kb+16..kb+23
//               with kb = 32*ks + 8*(L>>4).
//  B 32x16 f16: lane L -> col n0+(L&15); same K halves as A (kb..kb+7, kb+16..kb+23).
//  C/D f32:     VGPR r, lane L -> M = r + 8*(L>>4), N = (L&15).
// =====================================================================
__global__ __launch_bounds__(256) void gemm_wmma_kernel(
    const _Float16* __restrict__ pF16, const float* __restrict__ pwBias,
    const _Float16* __restrict__ crossF16, const float* __restrict__ nodeBias,
    float* __restrict__ out)
{
    __shared__ _Float16 Bs[128 * BSTRIDE];           // 34 KB

    const int tid  = threadIdx.x;
    const int lane = tid & 31;
    const int w    = __builtin_amdgcn_readfirstlane(tid >> 5);  // wave id
    const int n0   = blockIdx.x * 128;               // scalar
    const int m0   = blockIdx.y * 128 + w * 16;      // scalar
    const int hi   = lane >> 4;                      // 0 or 1
    const int lo   = lane & 15;

    const int aoff  = (m0 + lo) * DIM + hi * 8;      // into pF16 (global)
    const int bbase = lo * BSTRIDE + hi * 8;         // into Bs (LDS, f16 idx)

    const _Float16* gB = crossF16 + (size_t)n0 * DIM;

    v8f acc[8] = {v8f{}, v8f{}, v8f{}, v8f{}, v8f{}, v8f{}, v8f{}, v8f{}};

#pragma unroll
    for (int kk = 0; kk < 2; ++kk) {                 // two K-halves of 128
        __syncthreads();                             // done reading prev half
        // ---- stage B half: 128 rows x 128 f16; 8 x 16B chunks per thread
#pragma unroll
        for (int i = 0; i < 8; ++i) {
            const int c   = i * 256 + tid;           // 16B-chunk id
            const int row = c >> 4;                  // 16 chunks per row
            const int off = (c & 15) * 8;
            *(v8h*)&Bs[row * BSTRIDE + off] =
                *(const v8h*)(gB + (size_t)row * DIM + kk * 128 + off);
        }
        __syncthreads();                             // B half visible

        // ---- 4 k-steps of 32 against the staged half
#pragma unroll
        for (int ks = 0; ks < 4; ++ks) {
            const _Float16* ap = pF16 + aoff + (kk * 4 + ks) * 32;
            const v8h alo = *(const v8h*)(ap);
            const v8h ahi = *(const v8h*)(ap + 16);
            const v16h a = __builtin_shufflevector(alo, ahi,
                0,1,2,3,4,5,6,7,8,9,10,11,12,13,14,15);
#pragma unroll
            for (int j = 0; j < 8; ++j) {
                const _Float16* bp = &Bs[bbase + j * (16 * BSTRIDE) + ks * 32];
                const v8h blo = *(const v8h*)(bp);
                const v8h bhi = *(const v8h*)(bp + 16);
                const v16h b = __builtin_shufflevector(blo, bhi,
                    0,1,2,3,4,5,6,7,8,9,10,11,12,13,14,15);
                acc[j] = __builtin_amdgcn_wmma_f32_16x16x32_f16(
                    false, a, false, b, (short)0, acc[j], false, false);
            }
        }
    }

    // Epilogue: D(r, lane) -> out[m0 + r + 8*hi][n0 + 16*j + lo]
    const int mrow = m0 + hi * 8;
    float pw[8];
#pragma unroll
    for (int r = 0; r < 8; ++r) pw[r] = pwBias[mrow + r];

#pragma unroll
    for (int j = 0; j < 8; ++j) {
        const int ncol = n0 + j * 16 + lo;
        const float nb = nodeBias[ncol];             // in-bounds up to NPAD
        if (ncol < N_NODES) {                        // store guard only
#pragma unroll
            for (int r = 0; r < 8; ++r) {
                // Streaming output: non-temporal so the 205 MB result does
                // not evict the L2-resident crossF16 / pF16 working set.
                __builtin_nontemporal_store(
                    acc[j][r] + pw[r] + nb,
                    out + (size_t)(mrow + r) * N_NODES + ncol);
            }
        }
    }
}

// =====================================================================
// Launch
// =====================================================================
static inline uintptr_t align_up(uintptr_t p, uintptr_t a) {
    return (p + a - 1) & ~(a - 1);
}

extern "C" void kernel_launch(void* const* d_in, const int* in_sizes, int n_in,
                              void* d_out, int out_size, void* d_ws, size_t ws_size,
                              hipStream_t stream) {
    const float*     node_emb    = (const float*)d_in[0];     // 50000*256
    const float*     phrase_emb  = (const float*)d_in[1];     // 1024*256
    const float*     score_w     = (const float*)d_in[2];     // 2816
    const float*     score_b     = (const float*)d_in[3];     // 1
    const long long* neighbors   = (const long long*)d_in[4]; // 50000*4 (i64)
    const int*       nmask       = (const int*)d_in[5];       // 50000*4
    const int*       filter_mask = (const int*)d_in[6];       // 50000
    float*           out         = (float*)d_out;             // 1024*50000

    // workspace carve (~26.6 MB)
    uintptr_t p = align_up((uintptr_t)d_ws, 256);
    _Float16* pF16 = (_Float16*)p;        p = align_up(p + (size_t)N_PHRASES * DIM * 2, 256);
    float* pwBias  = (float*)p;           p = align_up(p + (size_t)N_PHRASES * 4, 256);
    float* invNorm = (float*)p;           p = align_up(p + (size_t)N_NODES * 4, 256);
    _Float16* crossF16 = (_Float16*)p;    p = align_up(p + (size_t)NPAD * DIM * 2, 256);
    float* nodeBias = (float*)p;          p = align_up(p + (size_t)NPAD * 4, 256);

    // 1) phrase prep: 1024 rows, 8 waves/block
    phrase_prep_kernel<<<N_PHRASES / 8, 256, 0, stream>>>(
        phrase_emb, score_w, score_b, pF16, pwBias);

    // 2a) node inverse norms: 50000 rows
    node_norm_kernel<<<(N_NODES + 7) / 8, 256, 0, stream>>>(node_emb, invNorm);

    // 2b) fused gather + cross/node_scalar over padded range
    node_feats_kernel<<<NPAD / 8, 256, 0, stream>>>(
        node_emb, score_w, neighbors, nmask, filter_mask, invNorm,
        crossF16, nodeBias);

    // 3) WMMA GEMM: 391 N-blocks x 8 M-blocks, 256 threads (8 waves) each
    dim3 grid(NGROUPS, MBLOCKS);
    gemm_wmma_kernel<<<grid, 256, 0, stream>>>(
        pF16, pwBias, crossF16, nodeBias, out);
}
